// NodeAggr_15470472201060
// MI455X (gfx1250) — matrix operand order repaired and checked by reading, not compiled, not run on hardware
//
#include <hip/hip_runtime.h>
#include <hip/hip_bf16.h>

typedef __attribute__((ext_vector_type(16))) __bf16 v16bf;
typedef __attribute__((ext_vector_type(8)))  float  v8f;

#define HID   128
#define AST   136   // LDS row stride (bf16 elems) for A buffer, padded
#define BST   136   // LDS row stride (bf16 elems) for B (W^T) buffer
#define TM    64    // nodes per block

__device__ __forceinline__ unsigned int pack_bf16(float a, float b) {
    unsigned int ua = __float_as_uint(a), ub = __float_as_uint(b);
    ua = (ua + 0x7FFFu + ((ua >> 16) & 1u)) >> 16;
    ub = (ub + 0x7FFFu + ((ub >> 16) & 1u)) >> 16;
    return ua | (ub << 16);
}
__device__ __forceinline__ unsigned short bf16_1(float a) {
    unsigned int ua = __float_as_uint(a);
    return (unsigned short)((ua + 0x7FFFu + ((ua >> 16) & 1u)) >> 16);
}

// tanh-gelu, branchless & overflow-safe:
// gelu(x) = x * sigmoid(2*sqrt(2/pi)*(x + 0.044715 x^3))
//         = x / (1 + exp(-1.5957691216 x - 0.0713548163 x^3))
__device__ __forceinline__ float gelu_tanh(float x) {
    float x2 = x * x;
    float s  = x * fmaf(-0.07135481627314696f, x2, -1.5957691216057308f);
    return x / (1.0f + __expf(s));
}

// A fragment (16x32 bf16, M=row per lane%16):
// element j holds K = kbase + (j<8?0:16) + 8*half + (j&7)   [ISA 7.12.2]
__device__ __forceinline__ v16bf load_frag_a(const unsigned short* buf, int row,
                                             int stride, int kbase, int half) {
    union { unsigned int u[8]; v16bf v; } f;
    const unsigned short* p = buf + row * stride + kbase + half * 8;
#pragma unroll
    for (int i = 0; i < 4; ++i) f.u[i]     = *(const unsigned int*)(p + 2 * i);
#pragma unroll
    for (int i = 0; i < 4; ++i) f.u[4 + i] = *(const unsigned int*)(p + 16 + 2 * i);
    return f.v;
}
// B fragment (32x16 bf16, N=col per lane%16), W stored transposed in LDS:
// element j holds K = kbase + (j<8?0:16) + 8*half + (j&7)
__device__ __forceinline__ v16bf load_frag_b(const unsigned short* buf, int col,
                                             int stride, int kbase, int half) {
    union { unsigned int u[8]; v16bf v; } f;
    const unsigned short* p = buf + col * stride + kbase + half * 8;
#pragma unroll
    for (int i = 0; i < 4; ++i) f.u[i]     = *(const unsigned int*)(p + 2 * i);
#pragma unroll
    for (int i = 0; i < 4; ++i) f.u[4 + i] = *(const unsigned int*)(p + 16 + 2 * i);
    return f.v;
}

__global__ void __launch_bounds__(256)
zero_f32_kernel(float* p, int n) {
    int i = blockIdx.x * blockDim.x + threadIdx.x;
    if (i < n) p[i] = 0.0f;
}

// Fused: m = gelu(gelu(x@W0+b0)@W1+b1); atomic scatter-add into agg[index[i]]
__global__ void __launch_bounds__(256)
msg_gemm_scatter_kernel(const float* __restrict__ x,
                        const int*   __restrict__ index,
                        const float* __restrict__ Wmsg,   // [2,128,128]
                        const float* __restrict__ bmsg,   // [2,128]
                        float*       __restrict__ agg,    // [4096,128]
                        int nNodes) {
    __shared__ unsigned short Abuf[TM * AST];
    __shared__ unsigned short Bbuf[HID * BST];
    __shared__ int idxl[TM];

    const int tid  = threadIdx.x;
    const int w    = tid >> 5;
    const int lane = tid & 31;
    const int half = lane >> 4;
    const int l16  = lane & 15;
    const int mt   = w & 3;         // 4 row tiles of 16
    const int nt   = w >> 2;        // 2 col groups of 64
    const int m0   = mt * 16;
    const long long blockRow = (long long)blockIdx.x * TM;

    // ---- stage x tile (f32 -> bf16), float4 (b128) global loads ----
#pragma unroll 2
    for (int p = tid; p < TM * (HID / 4); p += 256) {
        int r  = p >> 5;              // node row within tile
        int kq = (p & 31) << 2;       // feature quad
        float4 f = make_float4(0.f, 0.f, 0.f, 0.f);
        long long g = blockRow + r;
        if (g < nNodes)
            f = *(const float4*)(x + g * HID + kq);
        *(unsigned int*)(Abuf + r * AST + kq)     = pack_bf16(f.x, f.y);
        *(unsigned int*)(Abuf + r * AST + kq + 2) = pack_bf16(f.z, f.w);
    }
    // ---- stage W0^T (f32 -> bf16), Bbuf[n][k] = W0[k][n] ----
    const float* W0 = Wmsg;
#pragma unroll 4
    for (int p = tid; p < HID * (HID / 2); p += 256) {
        int n  = p >> 6;
        int kp = (p & 63) << 1;
        *(unsigned int*)(Bbuf + n * BST + kp) =
            pack_bf16(W0[kp * HID + n], W0[(kp + 1) * HID + n]);
    }
    if (tid < TM) {
        long long g = blockRow + tid;
        idxl[tid] = (g < nNodes) ? index[g] : -1;
    }
    __syncthreads();

    float res[4][8];

    // ================= layer 0 =================
#pragma unroll
    for (int t = 0; t < 4; ++t) {
        int n0 = nt * 64 + t * 16;
        v8f cc = {};
#pragma unroll
        for (int kb = 0; kb < HID; kb += 32) {
            v16bf a = load_frag_a(Abuf, m0 + l16, AST, kb, half);
            v16bf b = load_frag_b(Bbuf, n0 + l16, BST, kb, half);
            cc = __builtin_amdgcn_wmma_f32_16x16x32_bf16(false, a, false, b,
                                                         (short)0, cc, false, false);
        }
        float bias = bmsg[n0 + l16];
#pragma unroll
        for (int r = 0; r < 8; ++r) res[t][r] = gelu_tanh(cc[r] + bias);
    }
    __syncthreads();

    // write h1 (bf16) back to Abuf; stage W1^T
#pragma unroll
    for (int t = 0; t < 4; ++t) {
        int col = nt * 64 + t * 16 + l16;
#pragma unroll
        for (int r = 0; r < 8; ++r)
            Abuf[(m0 + r + 8 * half) * AST + col] = bf16_1(res[t][r]);
    }
    const float* W1 = Wmsg + HID * HID;
#pragma unroll 4
    for (int p = tid; p < HID * (HID / 2); p += 256) {
        int n  = p >> 6;
        int kp = (p & 63) << 1;
        *(unsigned int*)(Bbuf + n * BST + kp) =
            pack_bf16(W1[kp * HID + n], W1[(kp + 1) * HID + n]);
    }
    __syncthreads();

    // ================= layer 1 =================
#pragma unroll
    for (int t = 0; t < 4; ++t) {
        int n0 = nt * 64 + t * 16;
        v8f cc = {};
#pragma unroll
        for (int kb = 0; kb < HID; kb += 32) {
            v16bf a = load_frag_a(Abuf, m0 + l16, AST, kb, half);
            v16bf b = load_frag_b(Bbuf, n0 + l16, BST, kb, half);
            cc = __builtin_amdgcn_wmma_f32_16x16x32_bf16(false, a, false, b,
                                                         (short)0, cc, false, false);
        }
        float bias = bmsg[HID + n0 + l16];
#pragma unroll
        for (int r = 0; r < 8; ++r) res[t][r] = gelu_tanh(cc[r] + bias);
    }

    // ---- scatter-add into agg (L2-resident, 2MB) ----
#pragma unroll
    for (int t = 0; t < 4; ++t) {
        int col = nt * 64 + t * 16 + l16;
#pragma unroll
        for (int r = 0; r < 8; ++r) {
            int lrow = m0 + r + 8 * half;
            int idx  = idxl[lrow];
            if (idx >= 0)
                atomicAdd(agg + (long long)idx * HID + col, res[t][r]);
        }
    }
}

// out[j] = layernorm(r + agg[j]) * g + b, r = emb @ W_root + b_root
__global__ void __launch_bounds__(256)
pool_layernorm_kernel(const float* __restrict__ agg,
                      const float* __restrict__ emb,
                      const float* __restrict__ Wroot,
                      const float* __restrict__ broot,
                      const float* __restrict__ lng,
                      const float* __restrict__ lnb,
                      float* __restrict__ out, int nBins) {
    __shared__ float rv[HID];
    int tid = threadIdx.x;
    if (tid < HID) {
        float s = broot[tid];
#pragma unroll 4
        for (int k = 0; k < HID; ++k) s = fmaf(emb[k], Wroot[k * HID + tid], s);
        rv[tid] = s;
    }
    __syncthreads();

    int w = tid >> 5, lane = tid & 31;
    int row = blockIdx.x * 8 + w;
    if (row >= nBins) return;

    float h[4], s = 0.f, s2 = 0.f;
#pragma unroll
    for (int j = 0; j < 4; ++j) {
        int col = lane + 32 * j;
        h[j] = rv[col] + agg[(long long)row * HID + col];
        s += h[j];
        s2 = fmaf(h[j], h[j], s2);
    }
#pragma unroll
    for (int off = 16; off > 0; off >>= 1) {
        s  += __shfl_xor(s, off, 32);
        s2 += __shfl_xor(s2, off, 32);
    }
    float mu  = s * (1.0f / HID);
    float var = s2 * (1.0f / HID) - mu * mu;
    float inv = rsqrtf(var + 1e-5f);
#pragma unroll
    for (int j = 0; j < 4; ++j) {
        int col = lane + 32 * j;
        out[(long long)row * HID + col] = (h[j] - mu) * inv * lng[col] + lnb[col];
    }
}

extern "C" void kernel_launch(void* const* d_in, const int* in_sizes, int n_in,
                              void* d_out, int out_size, void* d_ws, size_t ws_size,
                              hipStream_t stream) {
    const float* x     = (const float*)d_in[0];
    const int*   index = (const int*)d_in[1];
    // d_in[2] = dim_size scalar on device; derive bins from out_size instead
    const float* emb   = (const float*)d_in[3];
    const float* Wmsg  = (const float*)d_in[4];
    const float* bmsg  = (const float*)d_in[5];
    const float* Wroot = (const float*)d_in[6];
    const float* broot = (const float*)d_in[7];
    const float* lng   = (const float*)d_in[8];
    const float* lnb   = (const float*)d_in[9];

    const int nNodes = in_sizes[0] / HID;
    const int nBins  = out_size / HID;     // 4096
    float* agg = (float*)d_ws;             // nBins*HID floats (2 MB)

    const int aggN = nBins * HID;
    zero_f32_kernel<<<(aggN + 255) / 256, 256, 0, stream>>>(agg, aggN);

    const int nBlocks = (nNodes + TM - 1) / TM;
    msg_gemm_scatter_kernel<<<nBlocks, 256, 0, stream>>>(x, index, Wmsg, bmsg,
                                                         agg, nNodes);

    pool_layernorm_kernel<<<(nBins + 7) / 8, 256, 0, stream>>>(agg, emb, Wroot,
                                                               broot, lng, lnb,
                                                               (float*)d_out, nBins);
}
